// MultiScaleClassifierFM_79362405696074
// MI455X (gfx1250) — matrix operand order repaired and checked by reading, not compiled, not run on hardware
//
#include <hip/hip_runtime.h>
#include <hip/hip_bf16.h>
#include <math.h>
#include <stdint.h>

using f16 = _Float16;
typedef __attribute__((ext_vector_type(16))) _Float16 v16h;
typedef __attribute__((ext_vector_type(8)))  _Float16 v8h;
typedef __attribute__((ext_vector_type(8)))  float    v8f;

#define WMMA_F16(Af, Bf, Cf) \
    __builtin_amdgcn_wmma_f32_16x16x32_f16(false, (Af), false, (Bf), (short)0, (Cf), false, false)

#define ASYNC_B128(ldsaddr, gaddr) \
    asm volatile("global_load_async_to_lds_b128 %0, %1, off" :: "v"(ldsaddr), "v"(gaddr) : "memory")
#define WAIT_ASYNC0() asm volatile("s_wait_asynccnt 0x0" ::: "memory")

static __device__ __forceinline__ float gelu_exact(float x) {
    return 0.5f * x * (1.0f + erff(x * 0.70710678118654752f));
}
static __device__ __forceinline__ unsigned lds_addr(const void* p) {
    return (unsigned)(uintptr_t)p;   // low 32 bits of LDS flat address = DS offset
}

// --------------- f32 KxN -> f16 NxK transpose (one-time weight prep) ----------
__global__ __launch_bounds__(256) void cvt_transpose_f16(const float* __restrict__ W,
                                                         f16* __restrict__ Wt,
                                                         int K, int N) {
    __shared__ float tile[32][33];
    const int k0 = blockIdx.y * 32, n0 = blockIdx.x * 32;
    const int tx = threadIdx.x & 31, ty = threadIdx.x >> 5;
    for (int r = ty; r < 32; r += 8)
        tile[r][tx] = W[(size_t)(k0 + r) * N + n0 + tx];
    __syncthreads();
    for (int r = ty; r < 32; r += 8)
        Wt[(size_t)(n0 + r) * K + k0 + tx] = (f16)tile[tx][r];
}

// ----------------------------- LayerNorm -> f16 -------------------------------
__global__ __launch_bounds__(256) void ln_f16(const float* __restrict__ X,
                                              const float* __restrict__ G,
                                              const float* __restrict__ Bt,
                                              f16* __restrict__ O, int D) {
    const int row = blockIdx.x;
    const float* x = X + (size_t)row * D;
    __shared__ float sred[256];
    const int t = threadIdx.x;

    float s = 0.f;
    for (int i = t; i < D; i += 256) s += x[i];
    sred[t] = s; __syncthreads();
    for (int st = 128; st > 0; st >>= 1) { if (t < st) sred[t] += sred[t + st]; __syncthreads(); }
    const float mean = sred[0] / (float)D;
    __syncthreads();

    float v = 0.f;
    for (int i = t; i < D; i += 256) { float d = x[i] - mean; v += d * d; }
    sred[t] = v; __syncthreads();
    for (int st = 128; st > 0; st >>= 1) { if (t < st) sred[t] += sred[t + st]; __syncthreads(); }
    const float inv = rsqrtf(sred[0] / (float)D + 1e-5f);
    __syncthreads();

    for (int i = t; i < D; i += 256)
        O[(size_t)row * D + i] = (f16)((x[i] - mean) * inv * G[i] + Bt[i]);
}

// ---------------- WMMA GEMM: C = act(A@W^T + bias) + resid --------------------
// A: MxK f16 row-major. Wt: NxK f16 (pre-transposed weight).
// 256 threads = 8 waves; block tile 128x64; K-step 64; wave tile 32x32.
// Tiles staged memory->LDS via global_load_async_to_lds_b128 (ASYNCcnt),
// double-buffered: one barrier per K-step, next stage overlaps current compute.
// LDS holds tiles in WMMA fragment layout -> fragment read = 2x ds_load_b128.
__global__ __launch_bounds__(256) void gemm_wmma(const f16* __restrict__ A,
                                                 const f16* __restrict__ Wt,
                                                 float* __restrict__ o32,
                                                 f16* __restrict__ o16,
                                                 const float* __restrict__ bias,
                                                 const float* __restrict__ resid,
                                                 int M, int N, int K, int act) {
    __shared__ v16h AsV[2][512];   // [buf][ksub(2)][subM(8)][lane(32)]
    __shared__ v16h BsV[2][256];   // [buf][ksub(2)][subN(4)][lane(32)]
    const int tid  = threadIdx.x;
    const int wave = tid >> 5, lane = tid & 31;
    const int m16  = lane & 15, hh = lane >> 4;
    const int bm = blockIdx.y * 128, bn = blockIdx.x * 64;
    const int wm2 = wave & 3;      // m-group (32 rows each)
    const int wn2 = wave >> 2;     // n-group (32 cols each)

    auto stage = [&](int k0, int p) {
        f16* Ad = (f16*)AsV[p];
        f16* Bd = (f16*)BsV[p];
#pragma unroll
        for (int i = 0; i < 4; ++i) {          // A: 128x64 halfs, 4 chunks/thread
            const int cid = i * 256 + tid;
            const int row = cid >> 3;
            const int c0  = (cid & 7) * 8;
            const int ks  = c0 >> 5, cc = c0 & 31;
            const int la  = (row & 15) + 16 * ((cc >> 3) & 1);
            const int ea  = (cc & 16) >> 1;
            const f16* gp = A + (size_t)(bm + row) * K + k0 + c0;
            ASYNC_B128(lds_addr(&Ad[(((ks * 8 + (row >> 4)) * 32 + la) << 4) + ea]), gp);
        }
#pragma unroll
        for (int i = 0; i < 2; ++i) {          // B: 64x64 halfs, 2 chunks/thread
            const int cid = i * 256 + tid;
            const int row = cid >> 3;
            const int c0  = (cid & 7) * 8;
            const int ks  = c0 >> 5, cc = c0 & 31;
            const int lb  = (row & 15) + 16 * ((cc >> 3) & 1);
            const int eb  = (cc & 16) >> 1;
            const f16* gp = Wt + (size_t)(bn + row) * K + k0 + c0;
            ASYNC_B128(lds_addr(&Bd[(((ks * 4 + (row >> 4)) * 32 + lb) << 4) + eb]), gp);
        }
    };

    v8f acc00 = {}, acc01 = {}, acc10 = {}, acc11 = {};
    stage(0, 0);
    int p = 0;
    for (int k0 = 0; k0 < K; k0 += 64, p ^= 1) {
        WAIT_ASYNC0();
        __syncthreads();
        if (k0 + 64 < K) stage(k0 + 64, p ^ 1);
        const v16h* Ab = AsV[p];
        const v16h* Bb = BsV[p];
#pragma unroll
        for (int ks = 0; ks < 2; ++ks) {
            v16h a0 = Ab[(ks * 8 + 2 * wm2 + 0) * 32 + lane];
            v16h a1 = Ab[(ks * 8 + 2 * wm2 + 1) * 32 + lane];
            v16h b0 = Bb[(ks * 4 + 2 * wn2 + 0) * 32 + lane];
            v16h b1 = Bb[(ks * 4 + 2 * wn2 + 1) * 32 + lane];
            acc00 = WMMA_F16(a0, b0, acc00);
            acc01 = WMMA_F16(a0, b1, acc01);
            acc10 = WMMA_F16(a1, b0, acc10);
            acc11 = WMMA_F16(a1, b1, acc11);
        }
    }
#pragma unroll
    for (int i = 0; i < 2; ++i) {
#pragma unroll
        for (int j = 0; j < 2; ++j) {
            const v8f acc = (i == 0) ? (j == 0 ? acc00 : acc01)
                                     : (j == 0 ? acc10 : acc11);
            const int col = bn + wn2 * 32 + j * 16 + m16;
            const float bs = bias ? bias[col] : 0.f;
#pragma unroll
            for (int r = 0; r < 8; ++r) {
                const int row = bm + wm2 * 32 + i * 16 + r + 8 * hh;
                float v = acc[r] + bs;
                if (act)   v = gelu_exact(v);
                if (resid) v += resid[(size_t)row * N + col];
                if (o32) o32[(size_t)row * N + col] = v;
                if (o16) o16[(size_t)row * N + col] = (f16)v;
            }
        }
    }
}

// ---------- fused flash attention with zero-sink softmax (DH = 64) ------------
// 4 waves/block; wave = 16 query rows of one (b, h). K tile staged via async
// LDS loads; V tile transposed into fragment layout; shared by all 4 waves.
__global__ __launch_bounds__(128) void attn_wmma(const f16* __restrict__ Qb,
                                                 const f16* __restrict__ Kb,
                                                 const f16* __restrict__ Vb,
                                                 f16* __restrict__ Ob,
                                                 int Nq, int Nk,
                                                 int qstr, int kstr, int vstr, int ostr,
                                                 int qoff0, int koff0, int voff0,
                                                 float scale, float wscale, int accum) {
    __shared__ v16h KsV[4 * 32];   // [ksub(2)][subJ(2)][lane(32)]
    __shared__ v16h VsV[4 * 32];   // [subN(4)][lane(32)]
    __shared__ v16h PsV[4 * 32];   // [wave(4)][lane(32)]
    f16* Ksf = (f16*)KsV;
    f16* Vsf = (f16*)VsV;
    f16* Psf = (f16*)PsV;
    const int tid  = threadIdx.x;
    const int wave = tid >> 5, lane = tid & 31;
    const int nn = lane & 15, hh = lane >> 4;
    const int h = blockIdx.y, b = blockIdx.z;
    const int q0 = (blockIdx.x * 4 + wave) * 16;
    const int qrow0 = b * Nq, krow0 = b * Nk;
    const f16* Qp = Qb + (size_t)(qrow0 + q0 + nn) * qstr + qoff0 + h * 64;
    const f16* Kp = Kb + (size_t)krow0 * kstr + koff0 + h * 64;
    const f16* Vp = Vb + (size_t)krow0 * vstr + voff0 + h * 64;
    const int ooff = h * 64;

    // Q fragments: 16x64 as two 16x32 A-fragments (16B vector loads)
    v8h q00 = *(const v8h*)(Qp + 8 * hh);
    v8h q01 = *(const v8h*)(Qp + 16 + 8 * hh);
    v8h q10 = *(const v8h*)(Qp + 32 + 8 * hh);
    v8h q11 = *(const v8h*)(Qp + 48 + 8 * hh);
    v16h qa0 = __builtin_shufflevector(q00, q01, 0,1,2,3,4,5,6,7,8,9,10,11,12,13,14,15);
    v16h qa1 = __builtin_shufflevector(q10, q11, 0,1,2,3,4,5,6,7,8,9,10,11,12,13,14,15);

    v8f zv = {};
    v8f o[4];
#pragma unroll
    for (int i = 0; i < 4; ++i) o[i] = zv;
    float mrow[8], lrow[8];
#pragma unroll
    for (int r = 0; r < 8; ++r) { mrow[r] = 0.0f; lrow[r] = 1.0f; }  // zero-sink

    const int lp = 16 * ((nn >> 3) & 1);   // P-frag lane offset from column
    const int ep = nn & 7;                 // P-frag element offset

    for (int j0 = 0; j0 < Nk; j0 += 32) {
        // ---- stage K (async b128 -> LDS frag layout) and V (transpose) ----
#pragma unroll
        for (int i = 0; i < 2; ++i) {
            const int cid = i * 128 + tid;     // 256 chunks of 8 halfs
            const int j  = cid >> 3;           // 0..31
            const int d0 = (cid & 7) * 8;      // 0..56
            const int ks = d0 >> 5, dd = d0 & 31;
            const int lk = (j & 15) + 16 * ((dd >> 3) & 1);
            const f16* gk = Kp + (size_t)(j0 + j) * kstr + d0;
            ASYNC_B128(lds_addr(&Ksf[(((ks * 2 + (j >> 4)) * 32 + lk) << 4) + ((dd & 16) >> 1)]), gk);

            v8h vv = *(const v8h*)(Vp + (size_t)(j0 + j) * vstr + d0);
            const int lv0 = 16 * ((j >> 3) & 1);
            const int ev  = (j & 7) + ((j & 16) >> 1);
#pragma unroll
            for (int u = 0; u < 8; ++u) {
                const int d = d0 + u;
                Vsf[((((d >> 4) * 32) + (d & 15) + lv0) << 4) + ev] = vv[u];
            }
        }
        WAIT_ASYNC0();
        __syncthreads();

        // ---- S tiles via WMMA from LDS fragments ----
        v8f s0 = zv, s1 = zv;
        {
            v16h k00 = KsV[(0 * 2 + 0) * 32 + lane];
            v16h k10 = KsV[(1 * 2 + 0) * 32 + lane];
            s0 = WMMA_F16(qa0, k00, s0);
            s0 = WMMA_F16(qa1, k10, s0);
            v16h k01 = KsV[(0 * 2 + 1) * 32 + lane];
            v16h k11 = KsV[(1 * 2 + 1) * 32 + lane];
            s1 = WMMA_F16(qa0, k01, s1);
            s1 = WMMA_F16(qa1, k11, s1);
        }

        // ---- online zero-sink softmax; emit P in A-fragment layout ----
#pragma unroll
        for (int r = 0; r < 8; ++r) {
            float x0 = s0[r] * scale;
            float x1 = s1[r] * scale;
            float bmax = fmaxf(x0, x1);
#pragma unroll
            for (int offm = 1; offm < 16; offm <<= 1)
                bmax = fmaxf(bmax, __shfl_xor(bmax, offm, 32));
            const float mnew = fmaxf(mrow[r], bmax);
            const float corr = __expf(mrow[r] - mnew);
            mrow[r] = mnew;
            const float p0 = __expf(x0 - mnew);
            const float p1 = __expf(x1 - mnew);
            float rs = p0 + p1;
#pragma unroll
            for (int offm = 1; offm < 16; offm <<= 1)
                rs += __shfl_xor(rs, offm, 32);
            lrow[r] = lrow[r] * corr + rs;
#pragma unroll
            for (int nb = 0; nb < 4; ++nb) o[nb][r] *= corr;
            const int plane = (r + 8 * hh) + lp;
            Psf[((wave * 32 + plane) << 4) + ep]     = (f16)p0;
            Psf[((wave * 32 + plane) << 4) + ep + 8] = (f16)p1;
        }
        __syncthreads();

        // ---- O += P @ V ----
        v16h pa = PsV[wave * 32 + lane];
#pragma unroll
        for (int nb = 0; nb < 4; ++nb) {
            v16h vb = VsV[nb * 32 + lane];
            o[nb] = WMMA_F16(pa, vb, o[nb]);
        }
        __syncthreads();
    }

#pragma unroll
    for (int r = 0; r < 8; ++r) {
        const float inv = wscale / lrow[r];
        const int row = qrow0 + q0 + r + 8 * hh;
#pragma unroll
        for (int nb = 0; nb < 4; ++nb) {
            const size_t idx = (size_t)row * ostr + ooff + nb * 16 + nn;
            float v = o[nb][r] * inv;
            if (accum) v += (float)Ob[idx];
            Ob[idx] = (f16)v;
        }
    }
}

// ------------------------------- orchestration --------------------------------
extern "C" void kernel_launch(void* const* d_in, const int* in_sizes, int n_in,
                              void* d_out, int out_size, void* d_ws, size_t ws_size,
                              hipStream_t stream) {
    (void)in_sizes; (void)n_in; (void)out_size; (void)ws_size;
    const int B = 4, NI = 1024, NC = 256, D = 768, INNER = 768, MFF = 3072, L = 2, NH = 12;
    const int RI = B * NI, RC = B * NC;

    const float* in_img = (const float*)d_in[0];
    const float* in_cli = (const float*)d_in[1];
    const float* ln1_g  = (const float*)d_in[2];
    const float* ln1_b  = (const float*)d_in[3];
    const float* w_qkv  = (const float*)d_in[4];
    const float* w_o    = (const float*)d_in[5];
    const float* b_o    = (const float*)d_in[6];
    const float* ln2_g  = (const float*)d_in[7];
    const float* ln2_b  = (const float*)d_in[8];
    const float* w_ff1  = (const float*)d_in[9];
    const float* b_ff1  = (const float*)d_in[10];
    const float* w_ff2  = (const float*)d_in[11];
    const float* b_ff2  = (const float*)d_in[12];
    const float* cn_i_g = (const float*)d_in[13];
    const float* cn_i_b = (const float*)d_in[14];
    const float* cn_c_g = (const float*)d_in[15];
    const float* cn_c_b = (const float*)d_in[16];
    const float* w_iq   = (const float*)d_in[17];
    const float* w_ik   = (const float*)d_in[18];
    const float* w_iv   = (const float*)d_in[19];
    const float* w_cq   = (const float*)d_in[20];
    const float* w_ck   = (const float*)d_in[21];
    const float* w_cv   = (const float*)d_in[22];
    const float* w_io   = (const float*)d_in[23];
    const float* b_io   = (const float*)d_in[24];
    const float* w_co   = (const float*)d_in[25];
    const float* b_co   = (const float*)d_in[26];
    const float* fn_i_g = (const float*)d_in[27];
    const float* fn_i_b = (const float*)d_in[28];
    const float* fn_c_g = (const float*)d_in[29];
    const float* fn_c_b = (const float*)d_in[30];
    const float* iff_w1 = (const float*)d_in[31];
    const float* iff_b1 = (const float*)d_in[32];
    const float* iff_w2 = (const float*)d_in[33];
    const float* iff_b2 = (const float*)d_in[34];
    const float* cff_w1 = (const float*)d_in[35];
    const float* cff_b1 = (const float*)d_in[36];
    const float* cff_w2 = (const float*)d_in[37];
    const float* cff_b2 = (const float*)d_in[38];

    char* wsb = (char*)d_ws;
    size_t woff = 0;
    auto alloc = [&](size_t bytes) -> void* {
        void* p = wsb + woff;
        woff += (bytes + 255) & ~(size_t)255;
        return p;
    };

    float* img = (float*)alloc((size_t)RI * D * 4);
    float* cli = (float*)alloc((size_t)RC * D * 4);
    f16* xn    = (f16*)alloc((size_t)RI * D * 2);
    f16* cxn   = (f16*)alloc((size_t)RC * D * 2);
    f16* qkv   = (f16*)alloc((size_t)RI * 3 * INNER * 2);   // reused as iq/ik/iv
    f16* att   = (f16*)alloc((size_t)RI * INNER * 2);
    f16* catt  = (f16*)alloc((size_t)RC * INNER * 2);
    f16* hbuf  = (f16*)alloc((size_t)RI * MFF * 2);
    f16* cqb   = (f16*)alloc((size_t)RC * INNER * 2);
    f16* ckb   = (f16*)alloc((size_t)RC * INNER * 2);
    f16* cvb   = (f16*)alloc((size_t)RC * INNER * 2);

    f16* wqkv16  = (f16*)alloc((size_t)L * D * 3 * INNER * 2);  // [3I][D] per layer
    f16* wo16    = (f16*)alloc((size_t)L * INNER * D * 2);      // [D][I] per layer
    f16* wff1_16 = (f16*)alloc((size_t)L * D * MFF * 2);        // [M][D] per layer
    f16* wff2_16 = (f16*)alloc((size_t)L * MFF * D * 2);        // [D][M] per layer
    f16* wiq16   = (f16*)alloc((size_t)D * INNER * 2);
    f16* wik16   = (f16*)alloc((size_t)D * INNER * 2);
    f16* wiv16   = (f16*)alloc((size_t)D * INNER * 2);
    f16* wcq16   = (f16*)alloc((size_t)D * INNER * 2);
    f16* wck16   = (f16*)alloc((size_t)D * INNER * 2);
    f16* wcv16   = (f16*)alloc((size_t)D * INNER * 2);
    f16* wio16   = (f16*)alloc((size_t)INNER * D * 2);
    f16* wco16   = (f16*)alloc((size_t)INNER * D * 2);
    f16* iw1_16  = (f16*)alloc((size_t)D * MFF * 2);
    f16* iw2_16  = (f16*)alloc((size_t)MFF * D * 2);
    f16* cw1_16  = (f16*)alloc((size_t)D * MFF * 2);
    f16* cw2_16  = (f16*)alloc((size_t)MFF * D * 2);

    // W is KxN f32 -> Wt is NxK f16
    auto cvtT = [&](const float* s, f16* dptr, int Kd, int Nd) {
        dim3 g(Nd / 32, Kd / 32);
        cvt_transpose_f16<<<g, 256, 0, stream>>>(s, dptr, Kd, Nd);
    };
    auto gemm = [&](const f16* A, const f16* Wt, float* o32, f16* o16,
                    const float* bias, const float* resid, int M, int N, int K, int act) {
        dim3 g(N / 64, M / 128);
        gemm_wmma<<<g, 256, 0, stream>>>(A, Wt, o32, o16, bias, resid, M, N, K, act);
    };
    auto attn = [&](const f16* Qb, const f16* Kb2, const f16* Vb, f16* Ob,
                    int Nq, int Nk, int qstr, int kstr, int vstr, int ostr,
                    int qoff, int koff, int voff, float wscale, int accum) {
        dim3 g(Nq / 64, NH, B);
        attn_wmma<<<g, 128, 0, stream>>>(Qb, Kb2, Vb, Ob, Nq, Nk, qstr, kstr, vstr, ostr,
                                         qoff, koff, voff, 0.125f, wscale, accum);
    };
    auto lnorm = [&](const float* X, const float* g, const float* bt, f16* O, int rows) {
        ln_f16<<<rows, 256, 0, stream>>>(X, g, bt, O, D);
    };

    // residual streams + transposed f16 weights (one-time)
    hipMemcpyAsync(img, in_img, (size_t)RI * D * 4, hipMemcpyDeviceToDevice, stream);
    hipMemcpyAsync(cli, in_cli, (size_t)RC * D * 4, hipMemcpyDeviceToDevice, stream);
    for (int l = 0; l < L; ++l) {
        cvtT(w_qkv + (size_t)l * D * 3 * INNER, wqkv16 + (size_t)l * 3 * INNER * D, D, 3 * INNER);
        cvtT(w_o   + (size_t)l * INNER * D,     wo16   + (size_t)l * D * INNER,     INNER, D);
        cvtT(w_ff1 + (size_t)l * D * MFF,       wff1_16 + (size_t)l * MFF * D,      D, MFF);
        cvtT(w_ff2 + (size_t)l * MFF * D,       wff2_16 + (size_t)l * D * MFF,      MFF, D);
    }
    cvtT(w_iq, wiq16, D, INNER);
    cvtT(w_ik, wik16, D, INNER);
    cvtT(w_iv, wiv16, D, INNER);
    cvtT(w_cq, wcq16, D, INNER);
    cvtT(w_ck, wck16, D, INNER);
    cvtT(w_cv, wcv16, D, INNER);
    cvtT(w_io, wio16, INNER, D);
    cvtT(w_co, wco16, INNER, D);
    cvtT(iff_w1, iw1_16, D, MFF);
    cvtT(iff_w2, iw2_16, MFF, D);
    cvtT(cff_w1, cw1_16, D, MFF);
    cvtT(cff_w2, cw2_16, MFF, D);

    // ---- self-attention transformer stack on image tokens ----
    for (int l = 0; l < L; ++l) {
        lnorm(img, ln1_g + l * D, ln1_b + l * D, xn, RI);
        gemm(xn, wqkv16 + (size_t)l * 3 * INNER * D, nullptr, qkv,
             nullptr, nullptr, RI, 3 * INNER, D, 0);
        attn(qkv, qkv, qkv, att, NI, NI, 3 * INNER, 3 * INNER, 3 * INNER, INNER,
             0, INNER, 2 * INNER, 1.0f, 0);
        gemm(att, wo16 + (size_t)l * D * INNER, img, nullptr,
             b_o + l * D, img, RI, D, INNER, 0);
        lnorm(img, ln2_g + l * D, ln2_b + l * D, xn, RI);
        gemm(xn, wff1_16 + (size_t)l * MFF * D, nullptr, hbuf,
             b_ff1 + l * MFF, nullptr, RI, MFF, D, 1 /*gelu*/);
        gemm(hbuf, wff2_16 + (size_t)l * D * MFF, img, nullptr,
             b_ff2 + l * D, img, RI, D, MFF, 0);
    }

    // ---- dual-stream cross + self attention fusion ----
    lnorm(img, cn_i_g, cn_i_b, xn, RI);
    lnorm(cli, cn_c_g, cn_c_b, cxn, RC);
    f16* iqb = qkv;
    f16* ikb = qkv + (size_t)RI * INNER;
    f16* ivb = qkv + (size_t)2 * RI * INNER;
    gemm(xn,  wiq16, nullptr, iqb, nullptr, nullptr, RI, INNER, D, 0);
    gemm(xn,  wik16, nullptr, ikb, nullptr, nullptr, RI, INNER, D, 0);
    gemm(xn,  wiv16, nullptr, ivb, nullptr, nullptr, RI, INNER, D, 0);
    gemm(cxn, wcq16, nullptr, cqb, nullptr, nullptr, RC, INNER, D, 0);
    gemm(cxn, wck16, nullptr, ckb, nullptr, nullptr, RC, INNER, D, 0);
    gemm(cxn, wcv16, nullptr, cvb, nullptr, nullptr, RC, INNER, D, 0);

    attn(iqb, ckb, cvb, att,  NI, NC, INNER, INNER, INNER, INNER, 0, 0, 0, 0.5f, 0);
    attn(iqb, ikb, ivb, att,  NI, NI, INNER, INNER, INNER, INNER, 0, 0, 0, 0.5f, 1);
    attn(cqb, ikb, ivb, catt, NC, NI, INNER, INNER, INNER, INNER, 0, 0, 0, 0.5f, 0);
    attn(cqb, ckb, cvb, catt, NC, NC, INNER, INNER, INNER, INNER, 0, 0, 0, 0.5f, 1);

    gemm(att,  wio16, img, nullptr, b_io, img, RI, D, INNER, 0);
    gemm(catt, wco16, cli, nullptr, b_co, cli, RC, D, INNER, 0);

    lnorm(img, fn_i_g, fn_i_b, xn, RI);
    gemm(xn, iw1_16, nullptr, hbuf, iff_b1, nullptr, RI, MFF, D, 1);
    gemm(hbuf, iw2_16, img, nullptr, iff_b2, img, RI, D, MFF, 0);
    lnorm(cli, fn_c_g, fn_c_b, cxn, RC);
    gemm(cxn, cw1_16, nullptr, hbuf, cff_b1, nullptr, RC, MFF, D, 1);
    gemm(hbuf, cw2_16, cli, nullptr, cff_b2, cli, RC, D, MFF, 0);

    // outputs: img then clinic, flat f32
    hipMemcpyAsync(d_out, img, (size_t)RI * D * 4, hipMemcpyDeviceToDevice, stream);
    hipMemcpyAsync((char*)d_out + (size_t)RI * D * 4, cli, (size_t)RC * D * 4,
                   hipMemcpyDeviceToDevice, stream);
}
